// ConvMultiHeadAttention_67448166417011
// MI455X (gfx1250) — matrix operand order repaired and checked by reading, not compiled, not run on hardware
//
#include <hip/hip_runtime.h>
#include <hip/hip_fp16.h>

typedef _Float16 f16;
typedef __attribute__((ext_vector_type(16))) _Float16 v16h;
typedef __attribute__((ext_vector_type(8)))  _Float16 v8h;
typedef __attribute__((ext_vector_type(8)))  float    v8f;
typedef __attribute__((ext_vector_type(4)))  float    v4f;

#define WMMA16(A, B, C) \
  __builtin_amdgcn_wmma_f32_16x16x32_f16(false, (A), false, (B), (short)0, (C), false, false)

union AF { v16h v; v8h h2[2]; };

#define S_  2048
#define D_  1024
#define H_  16
#define DH_ 64
#define B_  2

// ---------------------------------------------------------------------------
// Weight packing: wq_w/wk_w [C_out][C_in][3] -> f16 [C_out][k*1024 + C_in]
// ---------------------------------------------------------------------------
__global__ void pack_conv3_kernel(const float* __restrict__ w, f16* __restrict__ o, int n) {
  int i = blockIdx.x * blockDim.x + threadIdx.x;
  if (i >= n) return;
  int c   = i / 3072;
  int rem = i - c * 3072;
  int k   = rem >> 10;        // 0..2
  int ci  = rem & 1023;
  o[i] = (f16)w[(c * 1024 + ci) * 3 + k];
}

__global__ void pack_plain_kernel(const float* __restrict__ w, f16* __restrict__ o, int n) {
  int i = blockIdx.x * blockDim.x + threadIdx.x;
  if (i < n) o[i] = (f16)w[i];
}

// ---------------------------------------------------------------------------
// Projection GEMM (conv1d as GEMM).  Y[c][s] = bias[c] + sum_kk Wp[c][kk]*X[s+tap-pad][ci]
// mode 0: Q -> [b][h][s][d] (scaled 1/8)   mode 1: K -> [b][h][s][d]
// mode 2: V -> [b][h][d][s]
// One wave: 64 (M=channels) x 16 (N=positions) tile; K-loop in steps of 32.
// EDGE=true only for the two boundary s-tiles (conv halo); interior tiles run
// the branch-free fast path with prefetch of the next activation chunk.
// ---------------------------------------------------------------------------
template <bool EDGE>
static __device__ __forceinline__ void proj_kloop(
    const float* __restrict__ X, const f16* __restrict__ Wp,
    v8f acc[4], int Kt, int b, int mblk, int sn, int half, int lcol, int pad) {
  const int nkb = Kt >> 5;
  #pragma unroll 2
  for (int kb = 0; kb < nkb; ++kb) {
    const int kk  = kb << 5;
    const int tap = kk >> 10;                 // conv tap (0..2), uniform
    const int ci0 = (kk & 1023) + half * 16;  // this lane's 16-float chunk
    const int srow = sn + tap - pad;
    v16h bf;
    if (!EDGE || (srow >= 0 && srow < S_)) {
      const float* xp = X + ((size_t)b * S_ + srow) * D_ + ci0;
      v4f x0 = *(const v4f*)(xp);
      v4f x1 = *(const v4f*)(xp + 4);
      v4f x2 = *(const v4f*)(xp + 8);
      v4f x3 = *(const v4f*)(xp + 12);
      #pragma unroll
      for (int i = 0; i < 4; ++i) {
        bf[i]      = (f16)x0[i];
        bf[4 + i]  = (f16)x1[i];
        bf[8 + i]  = (f16)x2[i];
        bf[12 + i] = (f16)x3[i];
      }
    } else {
      #pragma unroll
      for (int i = 0; i < 16; ++i) bf[i] = (f16)0.0f;
    }
    if (!EDGE && (kb + 1 < nkb)) {
      const int kk2 = kk + 32;
      const float* xpn =
          X + ((size_t)b * S_ + sn + (kk2 >> 10) - pad) * D_ + ((kk2 & 1023) + half * 16);
      __builtin_prefetch(xpn, 0, 0);          // global_prefetch_b8
    }
    #pragma unroll
    for (int t = 0; t < 4; ++t) {
      const f16* ap = Wp + (size_t)(mblk * 64 + t * 16 + lcol) * Kt + kk;
      AF af;
      af.h2[0] = *(const v8h*)(ap + half * 16);
      af.h2[1] = *(const v8h*)(ap + half * 16 + 8);
      acc[t] = WMMA16(af.v, bf, acc[t]);
    }
  }
}

__global__ __launch_bounds__(256) void proj_kernel(
    const float* __restrict__ X,      // [B][S][D] fp32
    const f16*   __restrict__ Wp,     // [D][Kt] f16 packed
    const float* __restrict__ bias,   // [D]
    f16*         __restrict__ Y,
    int Kt, int mode, float scale) {
  const int wave = (blockIdx.x * blockDim.x + threadIdx.x) >> 5;
  const int lane = threadIdx.x & 31;
  const int half = lane >> 4;       // 0/1
  const int lcol = lane & 15;
  const int stile = wave & 127;
  const int mblk  = (wave >> 7) & 15;
  const int b     = wave >> 11;
  const int sn    = stile * 16 + lcol;     // this lane's N column (position)
  const int pad   = (Kt == 3072) ? 1 : 0;

  v8f acc[4];
  #pragma unroll
  for (int t = 0; t < 4; ++t) {
    const int m0 = mblk * 64 + t * 16 + half * 8;
    for (int r = 0; r < 8; ++r) acc[t][r] = bias[m0 + r];
  }

  if (pad && (stile == 0 || stile == 127))
    proj_kloop<true>(X, Wp, acc, Kt, b, mblk, sn, half, lcol, pad);
  else
    proj_kloop<false>(X, Wp, acc, Kt, b, mblk, sn, half, lcol, pad);

  #pragma unroll
  for (int t = 0; t < 4; ++t) {
    #pragma unroll
    for (int r = 0; r < 8; ++r) {
      const int m = mblk * 64 + t * 16 + half * 8 + r;  // channel
      const int head = m & 15;                          // head is FAST axis of channel
      const int dd   = m >> 4;
      const float v  = acc[t][r] * scale;
      size_t addr;
      if (mode == 2) addr = (((size_t)b * H_ + head) * DH_ + dd) * S_ + sn;
      else           addr = (((size_t)b * H_ + head) * S_ + sn) * DH_ + dd;
      Y[addr] = (f16)v;
    }
  }
}

// ---------------------------------------------------------------------------
// Flash attention: one wave per (b, h, 16-query tile). 32 iters of 64 keys:
// 8 score WMMAs + online softmax + LDS relayout + 8 PV WMMAs per iter.
// Q/K: [b][h][s][64] f16 ; V: [b][h][64][s] f16 ; out: [b][s][h*64+d] f16
// ---------------------------------------------------------------------------
__global__ __launch_bounds__(256) void attn_kernel(
    const f16* __restrict__ Qh, const f16* __restrict__ Kh,
    const f16* __restrict__ Vh, f16* __restrict__ O) {
  __shared__ __align__(16) f16 plds[8][16 * 64];   // per-wave P tile (16q x 64k)
  const int wave = (blockIdx.x * blockDim.x + threadIdx.x) >> 5;
  const int wv   = threadIdx.x >> 5;
  const int lane = threadIdx.x & 31;
  const int half = lane >> 4;
  const int lcol = lane & 15;
  const int qt = wave & 127;
  const int h  = (wave >> 7) & 15;
  const int b  = wave >> 11;

  const f16* Qb = Qh + (((size_t)b * H_ + h) * S_ + qt * 16) * DH_;
  const f16* Kb = Kh + (((size_t)b * H_ + h) * S_) * DH_;
  const f16* Vb = Vh + (((size_t)b * H_ + h) * DH_) * S_;

  AF qa[2];
  #pragma unroll
  for (int ks = 0; ks < 2; ++ks) {                  // K-dim (= d) in 2 steps of 32
    const f16* qp = Qb + (size_t)lcol * DH_ + ks * 32;
    qa[ks].h2[0] = *(const v8h*)(qp + half * 16);
    qa[ks].h2[1] = *(const v8h*)(qp + half * 16 + 8);
  }

  v8f o0 = {}, o1 = {}, o2 = {}, o3 = {};
  v8f* oacc[4] = { &o0, &o1, &o2, &o3 };
  float mrow[8], lrow[8];
  #pragma unroll
  for (int r = 0; r < 8; ++r) { mrow[r] = -3.0e38f; lrow[r] = 0.0f; }

  f16* pl = &plds[wv][0];
  for (int kt = 0; kt < 32; ++kt) {
    const int kb0 = kt * 64;
    // scores: four 16-key tiles, each K=64 (2 WMMAs)
    v8f s[4];
    #pragma unroll
    for (int t = 0; t < 4; ++t) {
      s[t] = {};
      #pragma unroll
      for (int ks = 0; ks < 2; ++ks) {
        const f16* kp = Kb + (size_t)(kb0 + t * 16 + lcol) * DH_ + ks * 32 + half * 16;
        AF kf;
        kf.h2[0] = *(const v8h*)(kp);
        kf.h2[1] = *(const v8h*)(kp + 8);
        s[t] = WMMA16(qa[ks].v, kf.v, s[t]);
      }
    }
    // online softmax over these 64 keys (rows: q = half*8 + r; cols across lcol)
    #pragma unroll
    for (int r = 0; r < 8; ++r) {
      float mx = fmaxf(fmaxf(s[0][r], s[1][r]), fmaxf(s[2][r], s[3][r]));
      #pragma unroll
      for (int off = 1; off < 16; off <<= 1) mx = fmaxf(mx, __shfl_xor(mx, off, 32));
      const float mnew = fmaxf(mrow[r], mx);
      const float corr = __expf(mrow[r] - mnew);
      float ps = 0.0f;
      #pragma unroll
      for (int t = 0; t < 4; ++t) {
        const float p = __expf(s[t][r] - mnew);
        s[t][r] = p;
        ps += p;
      }
      #pragma unroll
      for (int off = 1; off < 16; off <<= 1) ps += __shfl_xor(ps, off, 32);
      lrow[r] = lrow[r] * corr + ps;
      mrow[r] = mnew;
      o0[r] *= corr; o1[r] *= corr; o2[r] *= corr; o3[r] *= corr;
    }
    // D-frag -> A-frag relayout of P through LDS (wave-local, in-order DS pipe)
    #pragma unroll
    for (int t = 0; t < 4; ++t)
      #pragma unroll
      for (int r = 0; r < 8; ++r)
        pl[(half * 8 + r) * 64 + t * 16 + lcol] = (f16)s[t][r];
    asm volatile("s_wait_dscnt 0" ::: "memory");
    AF pa[2];
    #pragma unroll
    for (int ka = 0; ka < 2; ++ka) {            // P K-dim (keys) in 2 steps of 32
      pa[ka].h2[0] = *(const v8h*)(pl + lcol * 64 + ka * 32 + half * 16);
      pa[ka].h2[1] = *(const v8h*)(pl + lcol * 64 + ka * 32 + half * 16 + 8);
    }
    // O += P @ V  (4 N-tiles of d, 2 K-steps of 32 keys)
    #pragma unroll
    for (int nt = 0; nt < 4; ++nt) {
      #pragma unroll
      for (int ka = 0; ka < 2; ++ka) {
        const f16* vp = Vb + (size_t)(nt * 16 + lcol) * S_ + kb0 + ka * 32 + half * 16;
        AF vf;
        vf.h2[0] = *(const v8h*)(vp);
        vf.h2[1] = *(const v8h*)(vp + 8);
        *oacc[nt] = WMMA16(pa[ka].v, vf.v, *oacc[nt]);
      }
    }
  }

  // epilogue: normalize and store as [b][s][h*64 + d]
  #pragma unroll
  for (int r = 0; r < 8; ++r) {
    const float inv = 1.0f / lrow[r];
    const int qrow = qt * 16 + half * 8 + r;
    f16* op = O + ((size_t)b * S_ + qrow) * D_ + h * DH_;
    #pragma unroll
    for (int nt = 0; nt < 4; ++nt)
      op[nt * 16 + lcol] = (f16)((*oacc[nt])[r] * inv);
  }
}

// ---------------------------------------------------------------------------
// Output linear: out[b][s][j] = O[b][s][:] . wc[j][:] + bias[j]   (fp32 out)
// One wave: 16 (s) x 64 (j) tile, K=1024.
// ---------------------------------------------------------------------------
__global__ __launch_bounds__(256) void outproj_kernel(
    const f16* __restrict__ Ow, const f16* __restrict__ Wc,
    const float* __restrict__ bias, float* __restrict__ out) {
  const int wave = (blockIdx.x * blockDim.x + threadIdx.x) >> 5;
  const int lane = threadIdx.x & 31;
  const int half = lane >> 4;
  const int lcol = lane & 15;
  const int jblk  = wave & 15;
  const int stile = (wave >> 4) & 127;
  const int b     = wave >> 11;

  v8f acc0 = {}, acc1 = {}, acc2 = {}, acc3 = {};
  v8f* acc[4] = { &acc0, &acc1, &acc2, &acc3 };
  #pragma unroll 2
  for (int kb = 0; kb < 32; ++kb) {
    const int kk = kb << 5;
    const f16* ap = Ow + ((size_t)b * S_ + stile * 16 + lcol) * D_ + kk;
    AF af;
    af.h2[0] = *(const v8h*)(ap + half * 16);
    af.h2[1] = *(const v8h*)(ap + half * 16 + 8);
    #pragma unroll
    for (int nt = 0; nt < 4; ++nt) {
      const int j = jblk * 64 + nt * 16 + lcol;
      const f16* bp = Wc + (size_t)j * D_ + kk + half * 16;
      AF bf;
      bf.h2[0] = *(const v8h*)(bp);
      bf.h2[1] = *(const v8h*)(bp + 8);
      *acc[nt] = WMMA16(af.v, bf.v, *acc[nt]);
    }
  }
  #pragma unroll
  for (int nt = 0; nt < 4; ++nt) {
    const int j = jblk * 64 + nt * 16 + lcol;
    const float bj = bias[j];
    #pragma unroll
    for (int r = 0; r < 8; ++r) {
      const int srow = stile * 16 + half * 8 + r;
      out[((size_t)b * S_ + srow) * D_ + j] = (*acc[nt])[r] + bj;
    }
  }
}

// ---------------------------------------------------------------------------
extern "C" void kernel_launch(void* const* d_in, const int* in_sizes, int n_in,
                              void* d_out, int out_size, void* d_ws, size_t ws_size,
                              hipStream_t stream) {
  const float* q    = (const float*)d_in[0];
  const float* k    = (const float*)d_in[1];
  const float* v    = (const float*)d_in[2];
  const float* wq_w = (const float*)d_in[3];
  const float* wq_b = (const float*)d_in[4];
  const float* wk_w = (const float*)d_in[5];
  const float* wk_b = (const float*)d_in[6];
  const float* wv_w = (const float*)d_in[7];
  const float* wv_b = (const float*)d_in[8];
  const float* wc_w = (const float*)d_in[9];
  const float* wc_b = (const float*)d_in[10];
  float* out = (float*)d_out;

  char* ws = (char*)d_ws;
  f16* Wqp = (f16*)ws;  ws += (size_t)D_ * 3072 * sizeof(f16);   // 6 MB
  f16* Wkp = (f16*)ws;  ws += (size_t)D_ * 3072 * sizeof(f16);   // 6 MB
  f16* Wvp = (f16*)ws;  ws += (size_t)D_ * D_   * sizeof(f16);   // 2 MB
  f16* Wcp = (f16*)ws;  ws += (size_t)D_ * D_   * sizeof(f16);   // 2 MB
  f16* Qh  = (f16*)ws;  ws += (size_t)B_ * D_ * S_ * sizeof(f16); // 8 MB
  f16* Kh  = (f16*)ws;  ws += (size_t)B_ * D_ * S_ * sizeof(f16); // 8 MB
  f16* Vh  = (f16*)ws;  ws += (size_t)B_ * D_ * S_ * sizeof(f16); // 8 MB
  f16* Ow  = (f16*)ws;  ws += (size_t)B_ * D_ * S_ * sizeof(f16); // 8 MB

  // 1) pack weights to f16
  {
    int n3 = D_ * D_ * 3;
    pack_conv3_kernel<<<(n3 + 255) / 256, 256, 0, stream>>>(wq_w, Wqp, n3);
    pack_conv3_kernel<<<(n3 + 255) / 256, 256, 0, stream>>>(wk_w, Wkp, n3);
    int n1 = D_ * D_;
    pack_plain_kernel<<<(n1 + 255) / 256, 256, 0, stream>>>(wv_w, Wvp, n1);
    pack_plain_kernel<<<(n1 + 255) / 256, 256, 0, stream>>>(wc_w, Wcp, n1);
  }
  // 2) projections (conv as GEMM) -- 4096 waves each
  {
    dim3 grid(512), block(256);
    proj_kernel<<<grid, block, 0, stream>>>(q, Wqp, wq_b, Qh, 3072, 0, 0.125f); // 1/sqrt(64)
    proj_kernel<<<grid, block, 0, stream>>>(k, Wkp, wk_b, Kh, 3072, 1, 1.0f);
    proj_kernel<<<grid, block, 0, stream>>>(v, Wvp, wv_b, Vh, 1024, 2, 1.0f);
  }
  // 3) flash attention -- 4096 waves
  attn_kernel<<<512, 256, 0, stream>>>(Qh, Kh, Vh, Ow);
  // 4) output linear -- 4096 waves
  outproj_kernel<<<512, 256, 0, stream>>>(Ow, Wcp, wc_b, out);
}
